// StyleLossPass1_20246475833347
// MI455X (gfx1250) — matrix operand order repaired and checked by reading, not compiled, not run on hardware
//
#include <hip/hip_runtime.h>
#include <hip/hip_bf16.h>
#include <stdint.h>

// ---------------------------------------------------------------------------
// Problem constants (fixed by the reference: C=256, H=W=192, PS=3)
// ---------------------------------------------------------------------------
constexpr int CCH   = 256;             // channels
constexpr int WW    = 192;             // width
constexpr int NP    = 4096;            // number of patches (64*64)
constexpr int KP    = 2304;            // patch feature length C*PS*PS
constexpr int NPIX  = 36864;           // H*W  (= NP*9)

// scores-kernel blocking
constexpr int ROWS  = 32;              // content rows per block (2 row tiles)
constexpr int LDA   = KP + 8;          // LDS row stride in elements (+16B pad
                                       // -> 4624B stride, lanes land on
                                       // distinct bank quads for ds_load_b128)
constexpr size_t SMEM_A_BYTES   = (size_t)ROWS * LDA * 2;          // 147968
constexpr size_t SMEM_RED_OFF   = SMEM_A_BYTES;                    // 16B-aligned
constexpr size_t SMEM_BYTES     = SMEM_A_BYTES + 8 * ROWS * 4 * 2; // +sval+sidx

// workspace layout (bytes)
constexpr size_t OFF_CF   = 0;                         // bf16 [NP][KP]  content patches
constexpr size_t OFF_SF   = 18874368;                  // bf16 [NP][KP]  style patches
constexpr size_t OFF_IDX  = 37748736;                  // int  [NP]      argmax indices
constexpr size_t OFF_BM   = 37765120;                  // bf16 [C][NPIX] masked matched matrix
constexpr size_t OFF_GRAM = 56639488;                  // f32  [C][C]    unscaled gram

// WMMA types (probe-confirmed signatures)
typedef __attribute__((ext_vector_type(16))) __bf16 v16bf;
typedef __attribute__((ext_vector_type(8)))  float  v8f;

union V16U { v16bf v; uint4 u[2]; };

// ---------------------------------------------------------------------------
// Kernel 1: extract non-overlapping 3x3 patches into row-major bf16 [NP][KP].
// grid.y==0 -> content, grid.y==1 -> style.
// ---------------------------------------------------------------------------
__global__ void k_pack(const float* __restrict__ style,
                       const float* __restrict__ content,
                       __bf16* __restrict__ cfB,
                       __bf16* __restrict__ sfB) {
    const int flat = blockIdx.x * blockDim.x + threadIdx.x;   // [0, NP*KP)
    const float* src = blockIdx.y ? style : content;
    __bf16*      dst = blockIdx.y ? sfB   : cfB;

    const int k = flat % KP;
    const int p = flat / KP;
    const int c = k / 9;
    const int j = k % 9;
    const int gy = p >> 6, gx = p & 63;
    const int dy = j / 3,  dx = j % 3;
    const float val = src[(size_t)c * NPIX + (gy * 3 + dy) * WW + gx * 3 + dx];
    dst[(size_t)p * KP + k] = (__bf16)val;
}

// ---------------------------------------------------------------------------
// Kernel 2: fused scores GEMM (cf @ sf^T) + per-row argmax, LDS-blocked.
// 128 blocks x 256 threads (8 waves). Each block stages its 32-row A slab in
// LDS once (147KB of the 320KB/WGP). Per K-step: one streamed B fragment
// (global_load_b128 x2) + two A fragments (ds_load_b128 x4) feed TWO
// v_wmma_f32_16x16x32_bf16 -> 2x register reuse of B, ~zero A global traffic.
// Rolled loop with pointer-walking bases: each operand stream keeps its own
// address register with small immediate offsets (no per-iteration address
// VALU, nothing loop-invariant for LICM to hoist/spill).
// ---------------------------------------------------------------------------
__global__ void k_scores_argmax(const __bf16* __restrict__ cfB,
                                const __bf16* __restrict__ sfB,
                                int* __restrict__ idx) {
    extern __shared__ char smem[];
    __bf16* As   = (__bf16*)smem;
    float*  sval = (float*)(smem + SMEM_RED_OFF);
    int*    sidx = (int*)  (smem + SMEM_RED_OFF + 8 * ROWS * 4);

    const int lane  = threadIdx.x & 31;
    const int wave  = threadIdx.x >> 5;
    const int halve = lane >> 4;      // 0: lanes 0-15, 1: lanes 16-31
    const int l15   = lane & 15;
    const int rowbase = blockIdx.x * ROWS;

    // --- stage the 32xKP A slab into LDS (row-major, padded stride) ---
    {
        const __bf16* gsrc = cfB + (size_t)rowbase * KP;
        const int nvec = ROWS * (KP / 8);                 // uint4 chunks
        for (int i = threadIdx.x; i < nvec; i += 256) {
            const int r = i / (KP / 8);
            const int q = i % (KP / 8);
            *(uint4*)(As + (size_t)r * LDA + q * 8) =
                *(const uint4*)(gsrc + (size_t)r * KP + q * 8);
        }
    }
    __syncthreads();

    // A fragment bases (contiguous layout: lane half selects K base {0,16};
    // elems 0..15 -> K+0..15, matching B). Two independent
    // loop-invariant bases; the K loop walks them by +32 elements.
    const __bf16* a0p = As + (size_t)l15 * LDA + halve * 16;          // rows 0..15
    const __bf16* a1p = As + (size_t)(16 + l15) * LDA + halve * 16;   // rows 16..31

    float bestv[16];
    int   besti[16];
#pragma unroll
    for (int v = 0; v < 16; ++v) { bestv[v] = -3.4e38f; besti[v] = 0x7fffffff; }

    for (int ct = wave; ct < NP / 16; ct += 8) {
        const __bf16* pb  = sfB + (size_t)(ct * 16 + l15) * KP + halve * 16;
        const __bf16* pa0 = a0p;
        const __bf16* pa1 = a1p;

        v8f acc0 = {0.f, 0.f, 0.f, 0.f, 0.f, 0.f, 0.f, 0.f};
        v8f acc1 = {0.f, 0.f, 0.f, 0.f, 0.f, 0.f, 0.f, 0.f};
        for (int kit = 0; kit < KP / 32; ++kit) {
            V16U a0, a1, b;
            a0.u[0] = *(const uint4*)(pa0);              // ds_load_b128
            a0.u[1] = *(const uint4*)(pa0 + 8);
            a1.u[0] = *(const uint4*)(pa1);
            a1.u[1] = *(const uint4*)(pa1 + 8);
            b.u[0]  = *(const uint4*)(pb);               // global_load_b128
            b.u[1]  = *(const uint4*)(pb + 8);
            __builtin_prefetch(pb + 512, 0, 1);
            acc0 = __builtin_amdgcn_wmma_f32_16x16x32_bf16(
                       false, a0.v, false, b.v, (short)0, acc0, false, false);
            acc1 = __builtin_amdgcn_wmma_f32_16x16x32_bf16(
                       false, a1.v, false, b.v, (short)0, acc1, false, false);
            pa0 += 32; pa1 += 32; pb += 32;
        }

        // D layout: VGPR v, lanes 0-15 -> (m=v, n=lane); lanes 16-31 -> m=v+8.
#pragma unroll
        for (int t = 0; t < 2; ++t) {
#pragma unroll
            for (int v = 0; v < 8; ++v) {
                float val = t ? acc1[v] : acc0[v];
                int   ci  = ct * 16 + l15;
#pragma unroll
                for (int off = 8; off >= 1; off >>= 1) {
                    float ov = __shfl_xor(val, off, 16);
                    int   oc = __shfl_xor(ci,  off, 16);
                    if (ov > val || (ov == val && oc < ci)) { val = ov; ci = oc; }
                }
                const int r = t * 8 + v;
                if (val > bestv[r] || (val == bestv[r] && ci < besti[r])) {
                    bestv[r] = val; besti[r] = ci;
                }
            }
        }
    }

    if (l15 == 0) {        // lanes 0 and 16 publish rows {v+8*halve} per tile
#pragma unroll
        for (int t = 0; t < 2; ++t)
#pragma unroll
            for (int v = 0; v < 8; ++v) {
                const int row = t * 16 + v + 8 * halve;
                sval[wave * ROWS + row] = bestv[t * 8 + v];
                sidx[wave * ROWS + row] = besti[t * 8 + v];
            }
    }
    __syncthreads();
    if (threadIdx.x < ROWS) {             // fixed-order combine -> deterministic
        float bv = sval[threadIdx.x];
        int   bi = sidx[threadIdx.x];
        for (int w = 1; w < 8; ++w) {
            float v2 = sval[w * ROWS + threadIdx.x];
            int   i2 = sidx[w * ROWS + threadIdx.x];
            if (v2 > bv || (v2 == bv && i2 < bi)) { bv = v2; bi = i2; }
        }
        idx[rowbase + threadIdx.x] = bi;
    }
}

// ---------------------------------------------------------------------------
// Kernel 3: gather matched style patches, apply mask once per factor, build
// Bm[c][p*9+j] = sf[idx[p]][c*9+j] * mask[pixel].  One thread per (c,p).
// ---------------------------------------------------------------------------
__global__ void k_gather_mask(const __bf16* __restrict__ sfB,
                              const int* __restrict__ idx,
                              const float* __restrict__ mask,
                              __bf16* __restrict__ Bm) {
    const int flat = blockIdx.x * blockDim.x + threadIdx.x;   // [0, C*NP)
    const int p = flat & (NP - 1);
    const int c = flat >> 12;
    const int q = idx[p];
    const int gy = p >> 6, gx = p & 63;
    const __bf16* src = sfB + (size_t)q * KP + c * 9;
    __bf16*       dst = Bm  + (size_t)c * NPIX + p * 9;
#pragma unroll
    for (int j = 0; j < 9; ++j) {
        const int dy = j / 3, dx = j % 3;
        const float mk = mask[(gy * 3 + dy) * WW + gx * 3 + dx];
        dst[j] = (__bf16)((float)src[j] * mk);
    }
}

// ---------------------------------------------------------------------------
// Kernel 4: gram = Bm @ Bm^T (256x256, K = 36864) via WMMA bf16 -> f32.
// 128 blocks, each a 16x32 output slab: one A fragment feeds TWO WMMAs
// (register reuse). 8 waves split K (144 iters each); pointer-walking bases;
// deterministic fixed-order LDS combine of the 8 partial slabs.
// ---------------------------------------------------------------------------
__global__ void k_gram(const __bf16* __restrict__ Bm,
                       float* __restrict__ gram) {
    const int lane  = threadIdx.x & 31;
    const int wave  = threadIdx.x >> 5;
    const int halve = lane >> 4;
    const int l15   = lane & 15;
    const int ti  = blockIdx.x >> 3;        // output row tile   (0..15)
    const int tjp = blockIdx.x & 7;         // output col pair   (0..7)
    const int cols = tjp * 32;

    const int kk0 = wave * 32;
    const __bf16* pa  = Bm + (size_t)(ti * 16 + l15) * NPIX + halve * 16 + kk0;
    const __bf16* pb0 = Bm + (size_t)(cols + l15) * NPIX + halve * 16 + kk0;
    const __bf16* pb1 = Bm + (size_t)(cols + 16 + l15) * NPIX + halve * 16 + kk0;

    v8f acc0 = {0.f, 0.f, 0.f, 0.f, 0.f, 0.f, 0.f, 0.f};
    v8f acc1 = {0.f, 0.f, 0.f, 0.f, 0.f, 0.f, 0.f, 0.f};
    for (int kit = 0; kit < NPIX / 256; ++kit) {
        V16U a, b0, b1;
        a.u[0]  = *(const uint4*)(pa);
        a.u[1]  = *(const uint4*)(pa + 8);
        b0.u[0] = *(const uint4*)(pb0);
        b0.u[1] = *(const uint4*)(pb0 + 8);
        b1.u[0] = *(const uint4*)(pb1);
        b1.u[1] = *(const uint4*)(pb1 + 8);
        __builtin_prefetch(pa + 2048, 0, 1);
        acc0 = __builtin_amdgcn_wmma_f32_16x16x32_bf16(
                   false, a.v, false, b0.v, (short)0, acc0, false, false);
        acc1 = __builtin_amdgcn_wmma_f32_16x16x32_bf16(
                   false, a.v, false, b1.v, (short)0, acc1, false, false);
        pa += 256; pb0 += 256; pb1 += 256;
    }

    __shared__ float part[8][512];
#pragma unroll
    for (int v = 0; v < 8; ++v) {
        part[wave][v * 32 + lane]       = acc0[v];
        part[wave][256 + v * 32 + lane] = acc1[v];
    }
    __syncthreads();

#pragma unroll
    for (int s = 0; s < 2; ++s) {
        const int f = s * 256 + threadIdx.x;
        float sum = 0.f;
#pragma unroll
        for (int w = 0; w < 8; ++w) sum += part[w][f];   // fixed order
        const int v  = threadIdx.x >> 5;
        const int fl = threadIdx.x & 31;
        const int m  = v + 8 * (fl >> 4);
        const int n  = fl & 15;
        gram[(size_t)(ti * 16 + m) * CCH + cols + s * 16 + n] = sum;
    }
}

// ---------------------------------------------------------------------------
// Kernel 5: msum = sum(mask) (deterministic block reduction), then
// out = gram / (C * msum). Single block of 256 threads.
// ---------------------------------------------------------------------------
__global__ void k_finalize(const float* __restrict__ gram,
                           const float* __restrict__ mask,
                           float* __restrict__ out) {
    __shared__ float red[256];
    float s = 0.f;
    for (int m = threadIdx.x; m < NPIX; m += 256) s += mask[m];
    red[threadIdx.x] = s;
    __syncthreads();
    for (int off = 128; off > 0; off >>= 1) {
        if (threadIdx.x < off) red[threadIdx.x] += red[threadIdx.x + off];
        __syncthreads();
    }
    const float inv = 1.0f / (256.0f * red[0]);
    for (int e = threadIdx.x; e < CCH * CCH; e += 256) out[e] = gram[e] * inv;
}

// ---------------------------------------------------------------------------
// Launcher
// ---------------------------------------------------------------------------
extern "C" void kernel_launch(void* const* d_in, const int* in_sizes, int n_in,
                              void* d_out, int out_size, void* d_ws, size_t ws_size,
                              hipStream_t stream) {
    const float* style   = (const float*)d_in[0];
    const float* content = (const float*)d_in[1];
    const float* mask    = (const float*)d_in[2];
    float*       out     = (float*)d_out;

    char* base = (char*)d_ws;
    __bf16* cfB  = (__bf16*)(base + OFF_CF);
    __bf16* sfB  = (__bf16*)(base + OFF_SF);
    int*    idx  = (int*)   (base + OFF_IDX);
    __bf16* Bm   = (__bf16*)(base + OFF_BM);
    float*  gram = (float*) (base + OFF_GRAM);

    // 1) pack patches into bf16 [NP][KP]
    k_pack<<<dim3(NP * KP / 256, 2), 256, 0, stream>>>(style, content, cfB, sfB);

    // 2) fused scores GEMM + argmax (128 blocks, 147KB LDS A-slab each)
    k_scores_argmax<<<dim3(NP / ROWS), 256, SMEM_BYTES, stream>>>(cfB, sfB, idx);

    // 3) gather matched patches + mask (C*NP threads)
    k_gather_mask<<<dim3(CCH * NP / 256), 256, 0, stream>>>(sfB, idx, mask, Bm);

    // 4) masked gram via WMMA (128 slab blocks x 8 split-K waves)
    k_gram<<<dim3(128), 256, 0, stream>>>(Bm, gram);

    // 5) scale by 1/(C*sum(mask)) into d_out
    k_finalize<<<dim3(1), 256, 0, stream>>>(gram, mask, out);
}
